// get_mask_conv_modulated_40587440947381
// MI455X (gfx1250) — matrix operand order repaired and checked by reading, not compiled, not run on hardware
//
#include <hip/hip_runtime.h>
#include <hip/hip_bf16.h>
#include <math.h>
#include <stdint.h>

// ---------------- problem constants ----------------
#define B_     16
#define CIN_   512
#define COUT_  512
#define KSZ_   3
#define H_     32
#define W_     32
#define HW_    (H_ * W_)         // 1024
#define LDIM_  512
#define KKT_   (KSZ_ * KSZ_)     // 9
#define KTOT_  (CIN_ * KKT_)     // 4608
// equalized-lr scale: sqrt(2 / 4608) == 1/48 exactly
#define EQ_SCALE (1.0f / 48.0f)

// ---------------- tiling ----------------
#define MT_    128
#define NT_    256
#define KT_    32
#define LDSTR_ 40                // padded LDS row stride in halves (80B, 16B-aligned)

typedef _Float16 v8h  __attribute__((ext_vector_type(8)));
typedef _Float16 v16h __attribute__((ext_vector_type(16)));
typedef float    v8f  __attribute__((ext_vector_type(8)));

// ============================================================
// 1) style: s[b,i] = 1 + latent[b,:] . mlp_W[i,:] + mlp_b[i]
// ============================================================
__global__ __launch_bounds__(256)
void style_kernel(const float* __restrict__ latent,
                  const float* __restrict__ mlp_W,
                  const float* __restrict__ mlp_b,
                  float* __restrict__ s)
{
    int idx = blockIdx.x * 256 + threadIdx.x;      // 0 .. B*CIN-1
    int b = idx >> 9;
    int i = idx & (CIN_ - 1);
    const float* lat = latent + b * LDIM_;
    const float* wr  = mlp_W + (size_t)i * LDIM_;
    float acc = 0.0f;
    #pragma unroll 8
    for (int l = 0; l < LDIM_; ++l) acc = fmaf(lat[l], wr[l], acc);
    s[idx] = 1.0f + acc + mlp_b[i];
}

// ============================================================
// 2) prep: permuted-K weight staging.  k = r*512 + i
//    Wf16[o, r*512+i] = f16(scale * w[o,i,r] * relu6(mask[o,i])/6)
//    wsq[o,i] = sum_r (scale*w[o,i,r])^2   (pre-mask, for demod)
// ============================================================
__global__ __launch_bounds__(256)
void prep_kernel(const float* __restrict__ ori_w,
                 const float* __restrict__ wmask,
                 float* __restrict__ wsq,
                 _Float16* __restrict__ Wf16)
{
    int idx = blockIdx.x * 256 + threadIdx.x;      // idx = o*512+i
    float m = wmask[idx];
    m = fminf(fmaxf(m, 0.0f), 6.0f) * (1.0f / 6.0f);
    const float* wp = ori_w + (size_t)idx * KKT_;
    int o = idx >> 9;
    int i = idx & (CIN_ - 1);
    _Float16* dst = Wf16 + (size_t)o * KTOT_ + i;  // + r*512 per tap
    float sq = 0.0f;
    #pragma unroll
    for (int r = 0; r < KKT_; ++r) {
        float wv = wp[r] * EQ_SCALE;
        sq = fmaf(wv, wv, sq);
        dst[r * CIN_] = (_Float16)(wv * m);
    }
    wsq[idx] = sq;
}

// ============================================================
// 3) demod: d[b,o] = rsqrt( sum_i s[b,i]^2 * wsq[o,i] + 1e-5 )
// ============================================================
__global__ __launch_bounds__(256)
void demod_kernel(const float* __restrict__ s,
                  const float* __restrict__ wsq,
                  float* __restrict__ d)
{
    int idx = blockIdx.x * 256 + threadIdx.x;      // idx = b*512+o
    int b = idx >> 9;
    int o = idx & (COUT_ - 1);
    const float* sB = s   + b * CIN_;
    const float* wq = wsq + (size_t)o * CIN_;
    float acc = 1e-5f;
    #pragma unroll 8
    for (int i = 0; i < CIN_; ++i) {
        float sv = sB[i];
        acc = fmaf(sv * sv, wq[i], acc);
    }
    d[idx] = rsqrtf(acc);
}

// ============================================================
// 4) bias: nbias[o] = ori_bias[o] * relu6(bias_mask[o]) / 6
// ============================================================
__global__ __launch_bounds__(256)
void bias_kernel(const float* __restrict__ ori_bias,
                 const float* __restrict__ bias_mask,
                 float* __restrict__ nbias)
{
    int o = blockIdx.x * 256 + threadIdx.x;
    if (o < COUT_) {
        float m = fminf(fmaxf(bias_mask[o], 0.0f), 6.0f) * (1.0f / 6.0f);
        nbias[o] = ori_bias[o] * m;
    }
}

// ============================================================
// 5) conv: implicit-GEMM per batch via WMMA f32<=f16, permuted K.
//    Block tile 128M x 256N, 8 waves of 64x64 -> 16 WMMAs/wave/K-step.
//    A tile staged with GLOBAL_LOAD_ASYNC_TO_LDS_B128 (ASYNCcnt),
//    overlapped with the im2col B gather.
// ============================================================
__global__ __launch_bounds__(256)
void conv_wmma_kernel(const float* __restrict__ input,     // (B,CIN,H,W)
                      const _Float16* __restrict__ Wf16,   // (COUT, KTOT) permuted-K
                      const float* __restrict__ s,         // (B,CIN)
                      const float* __restrict__ d,         // (B,COUT)
                      const float* __restrict__ nbias,     // (COUT)
                      float* __restrict__ out)             // (B,COUT,H,W)
{
    __shared__ __align__(16) _Float16 ldsA[MT_ * LDSTR_];  // 10 KB
    __shared__ __align__(16) _Float16 ldsB[NT_ * LDSTR_];  // 20 KB

    const int tid   = threadIdx.x;
    const int lane  = tid & 31;
    const int wave  = tid >> 5;
    const int waveM = wave >> 2;           // 0..1  -> 64 rows each
    const int waveN = wave & 3;            // 0..3  -> 64 cols each
    const int mBlock = blockIdx.x * MT_;
    const int nBlock = blockIdx.y * NT_;
    const int b      = blockIdx.z;

    const float* inB = input + (size_t)b * CIN_ * HW_;
    const float* sB  = s + b * CIN_;

    // im2col fill: one full 32-K column per thread
    const int colFill = tid;               // 0..255
    const int p  = nBlock + colFill;
    const int py = p >> 5;
    const int px = p & 31;

    // A-stage assignment: 2 async b128 per thread
    const int aRow0 = tid >> 2;            // 0..63
    const int aCol0 = (tid & 3) * 8;
    const uint32_t ldsOffA0 = (uint32_t)(uintptr_t)(ldsA + aRow0 * LDSTR_ + aCol0);
    const uint32_t ldsOffA1 = ldsOffA0 + (uint32_t)(64 * LDSTR_ * 2);
    const uint32_t gRow0    = (uint32_t)(((mBlock + aRow0) * KTOT_ + aCol0) * 2);

    v8f acc[4][4];
    #pragma unroll
    for (int mf = 0; mf < 4; ++mf)
        #pragma unroll
        for (int nf = 0; nf < 4; ++nf)
            #pragma unroll
            for (int e = 0; e < 8; ++e) acc[mf][nf][e] = 0.0f;

    const int aRow  = lane & 15;
    const int khalf = (lane >> 4) * 8;     // 0 or 8
    const int bCol  = lane & 15;
    const int kbase = (lane >> 4) * 16;    // 0 or 16

    // outer loop over filter taps r (tap math hoisted), inner over channel chunks
    for (int r = 0; r < KKT_; ++r) {
        const int kh = r / 3;
        const int kw = r - kh * 3;
        const int y  = py + kh - 1;
        const int x  = px + kw - 1;
        const bool valid = ((unsigned)y < (unsigned)H_) && ((unsigned)x < (unsigned)W_);
        const float* src = inB + (y << 5) + x;   // only dereferenced if valid

        for (int ic = 0; ic < CIN_; ic += KT_) {
            const int kt = r * CIN_ + ic;
            __syncthreads();

            // ---- A tile: async global->LDS (no VGPR round-trip) ----
            {
                const uint32_t g0 = gRow0 + (uint32_t)(kt * 2);
                const uint32_t g1 = g0 + (uint32_t)(64 * KTOT_ * 2);
                asm volatile("global_load_async_to_lds_b128 %0, %1, %2"
                             :: "v"(ldsOffA0), "v"(g0), "s"(Wf16) : "memory");
                asm volatile("global_load_async_to_lds_b128 %0, %1, %2"
                             :: "v"(ldsOffA1), "v"(g1), "s"(Wf16) : "memory");
            }

            // ---- B tile: im2col gather (overlaps async A copy) ----
            {
                v8h h[4];
                if (valid) {
                    const float* sp = sB + ic;
                    const float* gp = src + (size_t)ic * HW_;
                    #pragma unroll
                    for (int q = 0; q < 4; ++q)
                        #pragma unroll
                        for (int j = 0; j < 8; ++j)
                            h[q][j] = (_Float16)(gp[(size_t)(q * 8 + j) * HW_] * sp[q * 8 + j]);
                } else {
                    #pragma unroll
                    for (int q = 0; q < 4; ++q)
                        #pragma unroll
                        for (int j = 0; j < 8; ++j) h[q][j] = (_Float16)0.f;
                }
                #pragma unroll
                for (int q = 0; q < 4; ++q)
                    *(v8h*)(ldsB + colFill * LDSTR_ + q * 8) = h[q];
            }

            asm volatile("s_wait_asynccnt 0" ::: "memory");
            __syncthreads();

            // ---- fragments (ISA VGPR layouts) + 16 WMMAs per wave ----
            v16h afr[4];
            #pragma unroll
            for (int mf = 0; mf < 4; ++mf) {
                const _Float16* ap = ldsA + (waveM * 64 + mf * 16 + aRow) * LDSTR_;
                v8h lo = *(const v8h*)(ap + khalf);
                v8h hi = *(const v8h*)(ap + 16 + khalf);
                #pragma unroll
                for (int j = 0; j < 8; ++j) { afr[mf][j] = lo[j]; afr[mf][8 + j] = hi[j]; }
            }
            #pragma unroll
            for (int nf = 0; nf < 4; ++nf) {
                const _Float16* bp = ldsB + (waveN * 64 + nf * 16 + bCol) * LDSTR_ + khalf;
                v8h lo = *(const v8h*)bp;
                v8h hi = *(const v8h*)(bp + 16);
                v16h bfr;
                #pragma unroll
                for (int j = 0; j < 8; ++j) { bfr[j] = lo[j]; bfr[8 + j] = hi[j]; }
                #pragma unroll
                for (int mf = 0; mf < 4; ++mf) {
                    acc[mf][nf] = __builtin_amdgcn_wmma_f32_16x16x32_f16(
                        false, afr[mf], false, bfr, (short)0, acc[mf][nf], false, false);
                }
            }
        }
    }

    // ---- epilogue: demod row-scale + bias, f32 store ----
    const int colOut  = lane & 15;
    const int rowHalf = (lane >> 4) * 8;
    const float* dB = d + b * COUT_;
    #pragma unroll
    for (int mf = 0; mf < 4; ++mf) {
        int oBase = mBlock + waveM * 64 + mf * 16 + rowHalf;
        float dv[8], bv[8];
        #pragma unroll
        for (int rr = 0; rr < 8; ++rr) { dv[rr] = dB[oBase + rr]; bv[rr] = nbias[oBase + rr]; }
        #pragma unroll
        for (int nf = 0; nf < 4; ++nf) {
            int pcol = nBlock + waveN * 64 + nf * 16 + colOut;
            #pragma unroll
            for (int rr = 0; rr < 8; ++rr) {
                float val = fmaf(acc[mf][nf][rr], dv[rr], bv[rr]);
                out[((size_t)b * COUT_ + (oBase + rr)) * HW_ + pcol] = val;
            }
        }
    }
}

// ============================================================
// host launcher
// ============================================================
extern "C" void kernel_launch(void* const* d_in, const int* in_sizes, int n_in,
                              void* d_out, int out_size, void* d_ws, size_t ws_size,
                              hipStream_t stream)
{
    const float* input     = (const float*)d_in[0];
    const float* latent    = (const float*)d_in[1];
    const float* ori_w     = (const float*)d_in[2];
    const float* ori_bias  = (const float*)d_in[3];
    const float* wmask     = (const float*)d_in[4];
    const float* bias_mask = (const float*)d_in[5];
    const float* mlp_W     = (const float*)d_in[6];
    const float* mlp_b     = (const float*)d_in[7];
    float* out = (float*)d_out;

    // workspace layout (256B aligned)
    char* ws = (char*)d_ws;
    const size_t off_s     = 0;                             // B*CIN f32      = 32 KB
    const size_t off_d     = off_s + 32768;                 // B*COUT f32     = 32 KB
    const size_t off_nb    = off_d + 32768;                 // COUT f32       = 2 KB
    const size_t off_wsq   = off_nb + 2048;                 // COUT*CIN f32   = 1 MB
    const size_t off_wf16  = off_wsq + (size_t)COUT_ * CIN_ * 4;   // COUT*KTOT f16 = 4.5 MB
    const size_t need      = off_wf16 + (size_t)COUT_ * KTOT_ * 2;
    if (ws_size < need) return;

    float*    s_buf  = (float*)(ws + off_s);
    float*    d_buf  = (float*)(ws + off_d);
    float*    nb_buf = (float*)(ws + off_nb);
    float*    wsq    = (float*)(ws + off_wsq);
    _Float16* Wf16   = (_Float16*)(ws + off_wf16);

    style_kernel<<<(B_ * CIN_) / 256, 256, 0, stream>>>(latent, mlp_W, mlp_b, s_buf);
    prep_kernel<<<(COUT_ * CIN_) / 256, 256, 0, stream>>>(ori_w, wmask, wsq, Wf16);
    demod_kernel<<<(B_ * COUT_) / 256, 256, 0, stream>>>(s_buf, wsq, d_buf);
    bias_kernel<<<(COUT_ + 255) / 256, 256, 0, stream>>>(ori_bias, bias_mask, nb_buf);

    dim3 grid(COUT_ / MT_, HW_ / NT_, B_);   // 4 x 4 x 16
    conv_wmma_kernel<<<grid, 256, 0, stream>>>(input, Wf16, s_buf, d_buf, nb_buf, out);
}
